// LSTMCellDecoder_75677323755595
// MI455X (gfx1250) — matrix operand order repaired and checked by reading, not compiled, not run on hardware
//
#include <hip/hip_runtime.h>

// Problem dimensions (fixed by the reference)
#define B_    64
#define L_    33
#define T_    32          // decode steps
#define ENC_  512
#define EMB_  256
#define H_    512
#define V_    32000
#define G4H_  (4*H_)      // 2048 gate width
#define KIN_  (ENC_+EMB_) // 768 concat(x,e) width
#define M_    (B_*T_)     // 2048 total (t,b) rows

typedef __bf16 bf16;
typedef __attribute__((ext_vector_type(8)))  bf16  v8bf;
typedef __attribute__((ext_vector_type(16))) bf16  v16bf;
typedef __attribute__((ext_vector_type(8)))  float v8f;
typedef __attribute__((ext_vector_type(4)))  unsigned int u32x4;
typedef __attribute__((ext_vector_type(8)))  int   i32x8;
typedef __attribute__((ext_vector_type(4)))  int   i32x4;

// ---------------------------------------------------------------------------
// WMMA helpers (CDNA5 wave32, 16x16x32 bf16 -> f32)
// A fragment: rows m0..m0+15, cols k0..k0+31 of row-major A[.][lda]
//   lane<16: M=lane,   K = {k0..k0+7, k0+16..k0+23}
//   lane>=16:M=lane-16,K = {k0+8..k0+15, k0+24..k0+31}
// ---------------------------------------------------------------------------
__device__ __forceinline__ v16bf frag_a(const bf16* A, int lda, int m0, int k0, int lane) {
    int row = m0 + (lane & 15);
    int kb  = k0 + ((lane & 16) ? 8 : 0);
    const v8bf* p = (const v8bf*)(A + (long)row * lda + kb);
    v8bf lo = p[0];          // K = kb .. kb+7
    v8bf hi = p[2];          // K = kb+16 .. kb+23
    return __builtin_shufflevector(lo, hi, 0,1,2,3,4,5,6,7,8,9,10,11,12,13,14,15);
}

// B fragment (K x N) sourced from row-major W[N][ldb] (i.e. B[k][n] = W[n][k]):
//   lane<16: N=lane,    K = {k0..k0+7, k0+16..k0+23}
//   lane>=16:N=lane-16, K = {k0+8..k0+15, k0+24..k0+31}
__device__ __forceinline__ v16bf frag_b(const bf16* W, int ldb, int n0, int k0, int lane) {
    int row = n0 + (lane & 15);
    int kb  = k0 + ((lane & 16) ? 8 : 0);
    const v8bf* p = (const v8bf*)(W + (long)row * ldb + kb);
    v8bf lo = p[0];
    v8bf hi = p[2];
    return __builtin_shufflevector(lo, hi, 0,1,2,3,4,5,6,7,8,9,10,11,12,13,14,15);
}

__device__ __forceinline__ v8f wmma_bf16(v16bf a, v16bf b, v8f c) {
    return __builtin_amdgcn_wmma_f32_16x16x32_bf16(false, a, false, b, (short)0, c, false, false);
}

// C/D element (r, lane): M = m0 + r + ((lane&16)?8:0), N = n0 + (lane&15)
__device__ __forceinline__ int cd_row(int m0, int r, int lane) { return m0 + r + ((lane & 16) ? 8 : 0); }
__device__ __forceinline__ int cd_col(int n0, int lane)        { return n0 + (lane & 15); }

__device__ __forceinline__ float sigf(float x) { return 1.0f / (1.0f + __expf(-x)); }

// ---------------------------------------------------------------------------
// 0) f32 -> bf16 weight conversion (grid-stride)
// ---------------------------------------------------------------------------
__global__ void k_cvt_bf16(const float* __restrict__ s, bf16* __restrict__ d, int n) {
    for (int i = blockIdx.x * blockDim.x + threadIdx.x; i < n; i += gridDim.x * blockDim.x)
        d[i] = (bf16)s[i];
}

// ---------------------------------------------------------------------------
// 1) stable descending argsort of target_lengths (B=64, O(B^2), one block)
// ---------------------------------------------------------------------------
__global__ void k_prep(const int* __restrict__ tlen, int* __restrict__ s_idx,
                       int* __restrict__ s_len, int* __restrict__ out_idx) {
    int i = threadIdx.x;
    if (i < B_) {
        int li = tlen[i], rank = 0;
        for (int j = 0; j < B_; ++j) {
            int lj = tlen[j];
            rank += (lj > li) || (lj == li && j < i);
        }
        s_idx[rank]   = i;
        s_len[rank]   = li - 1;     // decode steps for sorted row
        out_idx[rank] = i;          // sorted_indices output
    }
}

// ---------------------------------------------------------------------------
// 2) build A rows: Abuf[(t*64+b)][0:512] = x_sorted[b], [512:768] = embed[tgt_s[b][t]]
// ---------------------------------------------------------------------------
__global__ void k_build_A(const float* __restrict__ x, const float* __restrict__ embed,
                          const int* __restrict__ tgt, const int* __restrict__ s_idx,
                          bf16* __restrict__ Abuf) {
    const int n = M_ * KIN_;
    for (int idx = blockIdx.x * blockDim.x + threadIdx.x; idx < n; idx += gridDim.x * blockDim.x) {
        int m = idx / KIN_, k = idx - m * KIN_;
        int t = m >> 6, b = m & 63;
        int src = s_idx[b];
        float v;
        if (k < ENC_) v = x[src * ENC_ + k];
        else          v = embed[(long)tgt[src * L_ + t] * EMB_ + (k - ENC_)];
        Abuf[idx] = (bf16)v;
    }
}

// ---------------------------------------------------------------------------
// 3) initial state: h0 = x_s @ W_h0.T + b_h0 (-> bf16), c0 = x_s @ W_c0.T + b_c0
//    grid (32, 2): x = n-tile, y = {h,c}.  block = 128 (4 waves, one m-tile each)
// ---------------------------------------------------------------------------
__global__ void k_init_state(const bf16* __restrict__ Abuf,
                             const bf16* __restrict__ Wh0, const bf16* __restrict__ Wc0,
                             const float* __restrict__ bh0, const float* __restrict__ bc0,
                             bf16* __restrict__ hbf0, float* __restrict__ c) {
    const int lane = threadIdx.x & 31, wave = threadIdx.x >> 5;
    const int m0 = wave * 16, n0 = blockIdx.x * 16;
    const bf16*  W  = (blockIdx.y == 0) ? Wh0 : Wc0;
    const float* bb = (blockIdx.y == 0) ? bh0 : bc0;
    v8f acc = {};
    for (int k0 = 0; k0 < ENC_; k0 += 32) {
        v16bf a = frag_a(Abuf, KIN_, m0, k0, lane);   // x part lives in cols 0..511
        v16bf b = frag_b(W, ENC_, n0, k0, lane);
        acc = wmma_bf16(a, b, acc);
    }
    for (int r = 0; r < 8; ++r) {
        int row = cd_row(m0, r, lane), col = cd_col(n0, lane);
        float v = acc[r] + bb[col];
        if (blockIdx.y == 0) hbf0[row * H_ + col] = (bf16)v;
        else                 c[row * H_ + col]    = v;
    }
}

// ---------------------------------------------------------------------------
// 4) Gin[m][4H] = Abuf @ W_ih.T + b_ih + b_hh      (M=2048, N=2048, K=768)
//    grid (32, 32), block 128: wave -> m-tile, 4 n-accumulators
// ---------------------------------------------------------------------------
__global__ void k_gin(const bf16* __restrict__ Abuf, const bf16* __restrict__ Wih,
                      const float* __restrict__ bih, const float* __restrict__ bhh,
                      float* __restrict__ Gin) {
    const int lane = threadIdx.x & 31, wave = threadIdx.x >> 5;
    const int m0 = (blockIdx.y * 4 + wave) * 16;
    const int n0 = blockIdx.x * 64;
    v8f acc[4] = {};
    for (int k0 = 0; k0 < KIN_; k0 += 32) {
        v16bf a = frag_a(Abuf, KIN_, m0, k0, lane);
        #pragma unroll
        for (int i = 0; i < 4; ++i) {
            v16bf b = frag_b(Wih, KIN_, n0 + 16 * i, k0, lane);
            acc[i] = wmma_bf16(a, b, acc[i]);
        }
    }
    #pragma unroll
    for (int i = 0; i < 4; ++i)
        for (int r = 0; r < 8; ++r) {
            int row = cd_row(m0, r, lane), col = cd_col(n0 + 16 * i, lane);
            Gin[(long)row * G4H_ + col] = acc[i][r] + bih[col] + bhh[col];
        }
}

// ---------------------------------------------------------------------------
// 5) one decode step: g = Gin[t] + h @ W_hh.T, LSTM elementwise, mask, emit h_t
//    grid 32 (hidden chunks of 16), block 128 (4 waves -> 4 m-tiles of batch)
//    Each wave holds all 4 gate accumulators for its (batch x hidden) tile, so
//    the sigmoid/tanh update is fully thread-local.  hbf is double-buffered.
// ---------------------------------------------------------------------------
__global__ void k_step(int t, bf16* __restrict__ hbf, const bf16* __restrict__ Whh,
                       const float* __restrict__ Gin, float* __restrict__ c,
                       const int* __restrict__ s_len, bf16* __restrict__ Hall) {
    const int lane = threadIdx.x & 31, wave = threadIdx.x >> 5;
    const int m0 = wave * 16;                 // batch tile
    const int j0 = blockIdx.x * 16;           // hidden chunk
    const bf16* hprev = hbf + (size_t)(t & 1) * B_ * H_;
    bf16*       hnext = hbf + (size_t)((t + 1) & 1) * B_ * H_;

    v8f acc[4] = {};                          // gates i, f, g, o
    for (int k0 = 0; k0 < H_; k0 += 32) {
        v16bf a = frag_a(hprev, H_, m0, k0, lane);
        #pragma unroll
        for (int g = 0; g < 4; ++g) {
            v16bf b = frag_b(Whh, H_, g * H_ + j0, k0, lane);
            acc[g] = wmma_bf16(a, b, acc[g]);
        }
    }
    const long gbase = (long)(t * B_) * G4H_;
    #pragma unroll
    for (int r = 0; r < 8; ++r) {
        int row = cd_row(m0, r, lane);        // batch index (sorted order)
        int col = cd_col(j0, lane);           // hidden unit
        long gm = gbase + (long)row * G4H_;
        float gi = acc[0][r] + Gin[gm + 0 * H_ + col];
        float gf = acc[1][r] + Gin[gm + 1 * H_ + col];
        float gg = acc[2][r] + Gin[gm + 2 * H_ + col];
        float go = acc[3][r] + Gin[gm + 3 * H_ + col];
        float cp = c[row * H_ + col];
        float cn = sigf(gf) * cp + sigf(gi) * tanhf(gg);
        float hn = sigf(go) * tanhf(cn);
        bool act = s_len[row] > t;
        c[row * H_ + col]    = act ? cn : cp;
        hnext[row * H_ + col] = act ? (bf16)hn : hprev[row * H_ + col];
        Hall[(long)(t * B_ + row) * H_ + col] = (bf16)hn;   // masked in k_out
    }
}

// ---------------------------------------------------------------------------
// 6) output projection: preds = Hall @ W_out.T + b_out, masked, written [B,T,V]
//    M=2048, N=32000, K=512.  grid (500, 16), block 256 (8 waves -> 8 m-tiles),
//    4 n-accumulators per wave.
//    The 64x512 bf16 B-panel (64KB) is DMA'd once per block into LDS by the
//    Tensor Data Mover and shared by all 8 waves (8x cut in L2-side B traffic).
// ---------------------------------------------------------------------------
__global__ void __launch_bounds__(256)
k_out(const bf16* __restrict__ Hall, const bf16* __restrict__ Wout,
      const float* __restrict__ bout, const int* __restrict__ s_len,
      float* __restrict__ out) {
    __shared__ bf16 Bs[64 * H_];                        // 64 KB panel: rows n0..n0+63
    const int lane = threadIdx.x & 31, wave = threadIdx.x >> 5;
    const int m0 = (blockIdx.y * 8 + wave) * 16;
    const int n0 = blockIdx.x * 64;

    // --- TDM: DMA W_out[n0 .. n0+63][0..511] (contiguous 64KB) into LDS -----
    if (wave == 0) {
        unsigned long long ga = (unsigned long long)(uintptr_t)(Wout + (size_t)n0 * H_);
        unsigned ldsoff = (unsigned)(uintptr_t)(void*)Bs;   // LDS byte address = addr[31:0]
        u32x4 g0;
        g0.x = 1u;                                          // count=1, user descriptor
        g0.y = ldsoff;                                      // lds_addr [63:32]
        g0.z = (unsigned)(ga & 0xffffffffu);                // global_addr [95:64]
        g0.w = (unsigned)((ga >> 32) & 0x01ffffffu) | (2u << 30); // addr[56:32] | type=2
        i32x8 g1;
        g1[0] = (1 << 16);            // workgroup_mask=0 | data_size=1 (2 bytes)
        g1[1] = (H_ << 16);           // tensor_dim0[15:0]=512 (atomic_barrier_addr=0)
        g1[2] = (64 << 16);           // tensor_dim0[31:16]=0 | tensor_dim1[15:0]=64
        g1[3] = (int)((unsigned)H_ << 16); // tensor_dim1[31:16]=0 | tile_dim0=512
        g1[4] = 64;                   // tile_dim1=64 | tile_dim2=0
        g1[5] = H_;                   // tensor_dim0_stride[31:0]=512
        g1[6] = 0;                    // stride0[47:32]=0 | dim1_stride lo
        g1[7] = 0;
        i32x4 z4 = {};
        i32x8 z8 = {};
        __builtin_amdgcn_tensor_load_to_lds(g0, g1, z4, z4, z8, 0);
        __builtin_amdgcn_s_wait_tensorcnt(0);
    }
    __syncthreads();                                       // publish panel to all waves

    v8f acc[4] = {};
    for (int k0 = 0; k0 < H_; k0 += 32) {
        v16bf a = frag_a(Hall, H_, m0, k0, lane);          // own A rows from global/L2
        #pragma unroll
        for (int i = 0; i < 4; ++i) {
            v16bf b = frag_b(Bs, H_, 16 * i, k0, lane);    // shared panel from LDS
            acc[i] = wmma_bf16(a, b, acc[i]);
        }
    }
    #pragma unroll
    for (int i = 0; i < 4; ++i)
        for (int r = 0; r < 8; ++r) {
            int m  = cd_row(m0, r, lane);
            int t  = m >> 6, b = m & 63;
            int n  = cd_col(n0 + 16 * i, lane);
            float v = (s_len[b] > t) ? (acc[i][r] + bout[n]) : 0.0f;
            out[((long)b * T_ + t) * V_ + n] = v;          // [B, T, V]
        }
}

// ---------------------------------------------------------------------------
extern "C" void kernel_launch(void* const* d_in, const int* in_sizes, int n_in,
                              void* d_out, int out_size, void* d_ws, size_t ws_size,
                              hipStream_t stream) {
    const float* x      = (const float*)d_in[0];
    const int*   tgt    = (const int*)  d_in[1];
    const int*   tlen   = (const int*)  d_in[2];
    const float* embed  = (const float*)d_in[3];
    const float* W_h0   = (const float*)d_in[4];
    const float* b_h0   = (const float*)d_in[5];
    const float* W_c0   = (const float*)d_in[6];
    const float* b_c0   = (const float*)d_in[7];
    const float* W_ih   = (const float*)d_in[8];
    const float* W_hh   = (const float*)d_in[9];
    const float* b_ih   = (const float*)d_in[10];
    const float* b_hh   = (const float*)d_in[11];
    const float* W_out  = (const float*)d_in[12];
    const float* b_out  = (const float*)d_in[13];

    float* pred    = (float*)d_out;                       // [B, T, V]
    int*   out_idx = (int*)(pred + (size_t)B_ * T_ * V_); // sorted_indices

    // workspace carve (all offsets 256B-aligned)
    char* w = (char*)d_ws; size_t off = 0;
    auto carve = [&](size_t bytes) { void* p = w + off; off = (off + bytes + 255) & ~(size_t)255; return p; };
    int*   s_idx = (int*)  carve(B_ * sizeof(int));
    int*   s_len = (int*)  carve(B_ * sizeof(int));
    float* c     = (float*)carve((size_t)B_ * H_ * sizeof(float));
    bf16*  hbf   = (bf16*) carve((size_t)2 * B_ * H_ * sizeof(bf16));
    bf16*  Abuf  = (bf16*) carve((size_t)M_ * KIN_ * sizeof(bf16));
    float* Gin   = (float*)carve((size_t)M_ * G4H_ * sizeof(float));
    bf16*  Hall  = (bf16*) carve((size_t)M_ * H_ * sizeof(bf16));
    bf16*  Wihb  = (bf16*) carve((size_t)G4H_ * KIN_ * sizeof(bf16));
    bf16*  Whhb  = (bf16*) carve((size_t)G4H_ * H_ * sizeof(bf16));
    bf16*  Wh0b  = (bf16*) carve((size_t)H_ * ENC_ * sizeof(bf16));
    bf16*  Wc0b  = (bf16*) carve((size_t)H_ * ENC_ * sizeof(bf16));
    bf16*  Woutb = (bf16*) carve((size_t)V_ * H_ * sizeof(bf16));

    // weight conversions (f32 -> bf16)
    k_cvt_bf16<<<2048, 256, 0, stream>>>(W_ih,  Wihb,  G4H_ * KIN_);
    k_cvt_bf16<<<2048, 256, 0, stream>>>(W_hh,  Whhb,  G4H_ * H_);
    k_cvt_bf16<<<512,  256, 0, stream>>>(W_h0,  Wh0b,  H_ * ENC_);
    k_cvt_bf16<<<512,  256, 0, stream>>>(W_c0,  Wc0b,  H_ * ENC_);
    k_cvt_bf16<<<8192, 256, 0, stream>>>(W_out, Woutb, V_ * H_);

    // sort + gather + input projection (all parallel-in-t work hoisted)
    k_prep<<<1, 64, 0, stream>>>(tlen, s_idx, s_len, out_idx);
    k_build_A<<<3072, 256, 0, stream>>>(x, embed, tgt, s_idx, Abuf);
    k_init_state<<<dim3(32, 2), 128, 0, stream>>>(Abuf, Wh0b, Wc0b, b_h0, b_c0, hbf, c);
    k_gin<<<dim3(32, 32), 128, 0, stream>>>(Abuf, Wihb, b_ih, b_hh, Gin);

    // sequential recurrence: tiny per-step GEMM + fused LSTM elementwise
    for (int t = 0; t < T_; ++t)
        k_step<<<32, 128, 0, stream>>>(t, hbf, Whhb, Gin, c, s_len, Hall);

    // dominant batched output GEMM (67 GFLOP) on bf16 WMMA, TDM-staged B panel
    k_out<<<dim3(V_ / 64, 16), 256, 0, stream>>>(Hall, Woutb, b_out, s_len, pred);
}
